// InductionNetwork_57303453663498
// MI455X (gfx1250) — matrix operand order repaired and checked by reading, not compiled, not run on hardware
//
#include <hip/hip_runtime.h>

// CDNA5 / gfx1250 WMMA types (probe-confirmed signatures)
typedef __attribute__((ext_vector_type(16))) __bf16 v16bf;
typedef __attribute__((ext_vector_type(8)))  __bf16 v8bf;
typedef __attribute__((ext_vector_type(8)))  float  v8f;

#define CCAPS 256
#define KIN   64
#define HDIM  1024
#define LDA   1032             // padded bf16 row stride for x_hat in LDS

// ---- LDS layout (dynamic shared) ----
#define XHAT_OFF  0
#define XHAT_SZ   (64 * LDA * 2)                 // 132096 B
#define BPAN_ROW  80                             // 64B data + 16B pad (bank-conflict-free)
#define BPAN_BUF  (256 * BPAN_ROW)               // 20480 B per buffer
#define BPAN_OFF  (XHAT_OFF + XHAT_SZ)
#define BPAN_SZ   (2 * BPAN_BUF)                 // 40960 B (double buffer)
#define FSCR_OFF  (BPAN_OFF + BPAN_SZ)
#define SMEM_BYTES (FSCR_OFF + (HDIM + 64 + 64 + 1) * (int)sizeof(float))

#if __has_builtin(__builtin_amdgcn_sched_barrier)
#define SCHED_FENCE() __builtin_amdgcn_sched_barrier(0)
#else
#define SCHED_FENCE() asm volatile("" ::: "memory")
#endif

__device__ __forceinline__ float bf2f(unsigned int u16) {
    unsigned int x = u16 << 16;
    return __builtin_bit_cast(float, x);
}

// generic -> LDS byte offset (addrspace(3))
typedef __attribute__((address_space(3))) char* lds_ptr_t;
__device__ __forceinline__ unsigned lds_off_of(void* p) {
    return (unsigned)(unsigned long long)(lds_ptr_t)p;
}

// CDNA5 async copy: 16B per lane, global -> LDS, tracked by ASYNCcnt
__device__ __forceinline__ void async_cp_b128(unsigned lds_byte_off, const void* gsrc) {
    asm volatile("global_load_async_to_lds_b128 %0, %1, off"
                 :: "v"(lds_byte_off), "v"(gsrc) : "memory");
}
__device__ __forceinline__ void wait_asynccnt0() {
    asm volatile("s_wait_asynccnt 0x0" ::: "memory");
}

__global__ void cvt_w_bf16(const float* __restrict__ W, __bf16* __restrict__ Wb, int n) {
    int i = blockIdx.x * blockDim.x + threadIdx.x;
    if (i < n) Wb[i] = (__bf16)W[i];
}

__global__ __launch_bounds__(256)
void capsule_route_fused(const float* __restrict__ enc,     // [C, K, H] f32
                         const __bf16* __restrict__ Wb,     // [H, H] bf16 (row d, col h)
                         const int* __restrict__ iterPtr,   // scalar
                         float* __restrict__ out) {         // [C, H] f32
    extern __shared__ char smem_raw[];
    __bf16* xhat     = (__bf16*)(smem_raw + XHAT_OFF);      // [64][LDA] bf16
    float*  c_lds    = (float*)(smem_raw + FSCR_OFF);       // [HDIM]
    float*  b_lds    = c_lds + HDIM;                        // [64]
    float*  d_lds    = b_lds + 64;                          // [64]
    float*  norm_acc = d_lds + 64;                          // [1]
    const unsigned bpan_base = lds_off_of(smem_raw) + BPAN_OFF;

    const int tid  = threadIdx.x;
    const int lane = tid & 31;
    const int wave = tid >> 5;          // 8 waves (wave32)
    const int r    = lane & 15;         // A row-in-tile / B col-in-tile
    const int hi   = lane >> 4;         // lane half (WMMA bf16 layout)

    const int cIdx = blockIdx.x;
    const float* encC = enc + (size_t)cIdx * KIN * HDIM;

    // ---------- Phase 1: x_hat(64x1024) = enc_c @ W^T via v_wmma_f32_16x16x32_bf16 ----------
    // wave -> m-strip (wave>>1)*16, n-half (wave&1)*512; 4 groups of 8 N-tiles.
    // Per kt, a 256-row x 32-k bf16 B panel (the two active 128-col halves) is staged
    // into LDS with async-to-LDS copies, double buffered.
    const int mBase = (wave >> 1) * 16;
    const int half  = wave & 1;
    const float* arow = encC + (size_t)(mBase + r) * HDIM;

    // this thread's staging row (panel row p == tid)
    const int stage_n = (tid & 128) ? (512 + (tid & 127)) : tid;   // + g*128 below
    const unsigned stage_dst0 = bpan_base + (unsigned)tid * BPAN_ROW;

    for (int g = 0; g < 4; ++g) {
        const int nb0 = half * 512 + g * 128;
        const __bf16* stage_src_row = Wb + (size_t)(stage_n + g * 128) * HDIM;
        v8f acc[8] = {};

        // prologue: stage kt=0 into buffer 0
        {
            #pragma unroll
            for (int c2 = 0; c2 < 4; ++c2)
                async_cp_b128(stage_dst0 + c2 * 16, stage_src_row + c2 * 8);
            wait_asynccnt0();
            __syncthreads();
        }

        for (int kt = 0; kt < 32; ++kt) {
            const int k0 = kt * 32;

            // kick off async staging of the next panel into the other buffer
            if (kt < 31) {
                const __bf16* src = stage_src_row + (k0 + 32);
                const unsigned dst = stage_dst0 + (unsigned)((kt + 1) & 1) * BPAN_BUF;
                #pragma unroll
                for (int c2 = 0; c2 < 4; ++c2)
                    async_cp_b128(dst + c2 * 16, src + c2 * 8);
            }
            __builtin_prefetch(arow + k0 + 32, 0, 1);

            // raw A loads (global) ...
            float4 af0 = *(const float4*)(arow + k0 + hi * 8);
            float4 af1 = *(const float4*)(arow + k0 + hi * 8 + 4);
            float4 af2 = *(const float4*)(arow + k0 + 16 + hi * 8);
            float4 af3 = *(const float4*)(arow + k0 + 16 + hi * 8 + 4);

            // ... then ALL 16 B ds-loads into distinct registers.
            v16bf bt[8];
            #pragma unroll
            for (int t = 0; t < 8; ++t) {
                const int p = half * 128 + t * 16 + r;
                const char* rowp = smem_raw + BPAN_OFF + (kt & 1) * BPAN_BUF
                                 + p * BPAN_ROW + hi * 16;
                v8bf blo = *(const v8bf*)(rowp);
                v8bf bhi = *(const v8bf*)(rowp + 32);
                #pragma unroll
                for (int i = 0; i < 8; ++i) { bt[t][i] = blo[i]; bt[t][8 + i] = bhi[i]; }
            }

            // scheduling fence: nothing crosses. Forces all ds_loads to issue before
            // the WMMA stream, so waits stagger (in-order DS returns) instead of
            // a full s_wait_dscnt 0x0 LDS round trip per WMMA.
            SCHED_FENCE();

            // convert A (v_cvt_pk_bf16_f32)
            v16bf a;
            a[0]=(__bf16)af0.x;  a[1]=(__bf16)af0.y;  a[2]=(__bf16)af0.z;  a[3]=(__bf16)af0.w;
            a[4]=(__bf16)af1.x;  a[5]=(__bf16)af1.y;  a[6]=(__bf16)af1.z;  a[7]=(__bf16)af1.w;
            a[8]=(__bf16)af2.x;  a[9]=(__bf16)af2.y;  a[10]=(__bf16)af2.z; a[11]=(__bf16)af2.w;
            a[12]=(__bf16)af3.x; a[13]=(__bf16)af3.y; a[14]=(__bf16)af3.z; a[15]=(__bf16)af3.w;

            #pragma unroll
            for (int t = 0; t < 8; ++t) {
                acc[t] = __builtin_amdgcn_wmma_f32_16x16x32_bf16(
                    false, a, false, bt[t], (short)0, acc[t], false, false);
            }

            if (kt < 31) {
                wait_asynccnt0();      // next panel landed (this wave's copies)
                __syncthreads();       // visible to all; cur buffer free for reuse
            }
        }

        // D layout: VGPR j -> row mBase + hi*8 + j, lane -> col; store bf16 x_hat to LDS
        #pragma unroll
        for (int t = 0; t < 8; ++t) {
            const int col = nb0 + t * 16 + r;
            #pragma unroll
            for (int j = 0; j < 8; ++j) {
                const int row = mBase + hi * 8 + j;
                xhat[row * LDA + col] = (__bf16)acc[t][j];
            }
        }
        __syncthreads();   // all reads of both panel buffers done before next g restages
    }

    if (tid < 64) b_lds[tid] = 0.0f;
    __syncthreads();

    // ---------- Phase 2: dynamic routing entirely in LDS ----------
    const int iters = iterPtr[0];
    const int h4 = tid * 4;   // each thread owns 4 H-columns

    for (int it = 0; it < iters; ++it) {
        if (tid == 0) {        // softmax over K=64 (tiny)
            float mx = b_lds[0];
            for (int k = 1; k < 64; ++k) mx = fmaxf(mx, b_lds[k]);
            float s = 0.0f;
            for (int k = 0; k < 64; ++k) { float e = __expf(b_lds[k] - mx); d_lds[k] = e; s += e; }
            float inv = 1.0f / s;
            for (int k = 0; k < 64; ++k) d_lds[k] *= inv;
            *norm_acc = 0.0f;
        }
        __syncthreads();

        // c_hat[h] = sum_k d[k] * x_hat[k,h]
        float ch0 = 0.f, ch1 = 0.f, ch2 = 0.f, ch3 = 0.f;
        for (int k = 0; k < 64; ++k) {
            const float dk = d_lds[k];
            const uint2 raw = *(const uint2*)(xhat + k * LDA + h4);
            ch0 += dk * bf2f(raw.x & 0xffffu);
            ch1 += dk * bf2f(raw.x >> 16);
            ch2 += dk * bf2f(raw.y & 0xffffu);
            ch3 += dk * bf2f(raw.y >> 16);
        }
        atomicAdd(norm_acc, ch0 * ch0 + ch1 * ch1 + ch2 * ch2 + ch3 * ch3);
        __syncthreads();

        // squash: scale = (n/(1+n)) / sqrt(n)
        const float nrm = *norm_acc;
        const float scale = (nrm / (1.0f + nrm)) * rsqrtf(nrm);
        const float cv0 = scale * ch0, cv1 = scale * ch1, cv2 = scale * ch2, cv3 = scale * ch3;
        c_lds[h4 + 0] = cv0; c_lds[h4 + 1] = cv1; c_lds[h4 + 2] = cv2; c_lds[h4 + 3] = cv3;
        if (it == iters - 1) {
            *(float4*)(out + (size_t)cIdx * HDIM + h4) = make_float4(cv0, cv1, cv2, cv3);
        }
        __syncthreads();

        if (it < iters - 1) {
            // b[k] += sum_h x_hat[k,h] * c[h]
            const int k = tid & 63, q = tid >> 6;
            const __bf16* xr = xhat + k * LDA + q * 256;
            const float* cp = c_lds + q * 256;
            float p = 0.0f;
            for (int h = 0; h < 256; h += 4) {
                const uint2 raw = *(const uint2*)(xr + h);
                p += cp[h + 0] * bf2f(raw.x & 0xffffu);
                p += cp[h + 1] * bf2f(raw.x >> 16);
                p += cp[h + 2] * bf2f(raw.y & 0xffffu);
                p += cp[h + 3] * bf2f(raw.y >> 16);
            }
            atomicAdd(&b_lds[k], p);   // ds_add_f32
            __syncthreads();
        }
    }
}

extern "C" void kernel_launch(void* const* d_in, const int* in_sizes, int n_in,
                              void* d_out, int out_size, void* d_ws, size_t ws_size,
                              hipStream_t stream) {
    const float* enc     = (const float*)d_in[0];   // [256, 64, 1024] f32
    const float* W       = (const float*)d_in[1];   // [1024, 1024] f32
    const int*   iterPtr = (const int*)d_in[2];     // scalar (3)
    float* out = (float*)d_out;                     // [256, 1024] f32
    __bf16* Wb = (__bf16*)d_ws;                     // 2 MB scratch for bf16 W

    (void)hipFuncSetAttribute((const void*)capsule_route_fused,
                              hipFuncAttributeMaxDynamicSharedMemorySize, SMEM_BYTES);

    cvt_w_bf16<<<(HDIM * HDIM + 255) / 256, 256, 0, stream>>>(W, Wb, HDIM * HDIM);
    capsule_route_fused<<<CCAPS, 256, SMEM_BYTES, stream>>>(enc, Wb, iterPtr, out);
}
